// StackPointnetFPModule_5016521802209
// MI455X (gfx1250) — matrix operand order repaired and checked by reading, not compiled, not run on hardware
//
#include <hip/hip_runtime.h>
#include <hip/hip_bf16.h>

typedef __bf16 bf16;
typedef __attribute__((ext_vector_type(16))) __bf16 v16bf;
typedef __attribute__((ext_vector_type(8)))  float  v8f;

#define NN_TILE 512

// ---------------------------------------------------------------------------
// Kernel 1: ragged brute-force 3-NN + inverse-distance weights
// One thread per unknown point; known points staged through LDS in tiles,
// next tile prefetched (global_prefetch_b8).
// ---------------------------------------------------------------------------
__global__ void nn_kernel(const float* __restrict__ unknown,
                          const int* __restrict__ ucnt, int B,
                          const float* __restrict__ known,
                          const int* __restrict__ kcnt,
                          int N, int M,
                          int* __restrict__ out_idx,
                          float* __restrict__ out_w)
{
    __shared__ float skx[NN_TILE], sky[NN_TILE], skz[NN_TILE];
    __shared__ int sMin, sMax;

    int n = blockIdx.x * blockDim.x + threadIdx.x;

    // batch membership -> known index range [ks, ke)
    int ks = 0, ke = 0;
    {
        int ucum = 0, kcum = 0;
        for (int b = 0; b < B; ++b) {
            int un = ucnt[b], kn = kcnt[b];
            if (n >= ucum && n < ucum + un) { ks = kcum; ke = kcum + kn; }
            ucum += un; kcum += kn;
        }
    }
    if (threadIdx.x == 0) { sMin = 0x7fffffff; sMax = 0; }
    __syncthreads();
    atomicMin(&sMin, ks);
    atomicMax(&sMax, ke);
    __syncthreads();
    const int tStart = sMin, tEnd = sMax;   // block-uniform

    float ux = 0.f, uy = 0.f, uz = 0.f;
    if (n < N) { ux = unknown[n * 3 + 0]; uy = unknown[n * 3 + 1]; uz = unknown[n * 3 + 2]; }

    const float INF = 3.0e38f;
    float d0 = INF, d1 = INF, d2 = INF;
    int   i0 = 0,  i1 = 0,  i2 = 0;

    for (int t0 = tStart; t0 < tEnd; t0 += NN_TILE) {
        int cnt = min(NN_TILE, tEnd - t0);
        __syncthreads();
        // prefetch next tile while staging this one
        {
            int pj = t0 + NN_TILE + (int)threadIdx.x;
            if (pj < tEnd) __builtin_prefetch(&known[(size_t)pj * 3], 0, 0);
        }
        for (int j = threadIdx.x; j < cnt; j += blockDim.x) {
            skx[j] = known[(size_t)(t0 + j) * 3 + 0];
            sky[j] = known[(size_t)(t0 + j) * 3 + 1];
            skz[j] = known[(size_t)(t0 + j) * 3 + 2];
        }
        __syncthreads();
        int lo = max(ks, t0) - t0;
        int hi = min(ke, t0 + cnt) - t0;
        for (int j = lo; j < hi; ++j) {
            float dx = ux - skx[j], dy = uy - sky[j], dz = uz - skz[j];
            float d = dx * dx + dy * dy + dz * dz;
            int gi = t0 + j;
            if (d < d2) {
                if (d < d1) {
                    if (d < d0) { d2 = d1; i2 = i1; d1 = d0; i1 = i0; d0 = d; i0 = gi; }
                    else        { d2 = d1; i2 = i1; d1 = d;  i1 = gi; }
                } else          { d2 = d;  i2 = gi; }
            }
        }
    }

    if (n < N) {
        float r0 = 1.f / (d0 + 1e-8f);
        float r1 = 1.f / (d1 + 1e-8f);
        float r2 = 1.f / (d2 + 1e-8f);
        float s = 1.f / (r0 + r1 + r2);
        out_idx[n * 3 + 0] = i0; out_idx[n * 3 + 1] = i1; out_idx[n * 3 + 2] = i2;
        out_w[n * 3 + 0] = r0 * s; out_w[n * 3 + 1] = r1 * s; out_w[n * 3 + 2] = r2 * s;
    }
}

// ---------------------------------------------------------------------------
// Kernel 2: gather + weighted interpolation + concat, emit bf16 GEMM input
// ---------------------------------------------------------------------------
__global__ void interp_concat_kernel(const int* __restrict__ idx,
                                     const float* __restrict__ wgt,
                                     const float* __restrict__ kfeat,
                                     const float* __restrict__ ufeat,
                                     bf16* __restrict__ X,
                                     int C1, int C2)
{
    int n = blockIdx.x;
    int i0 = idx[n * 3 + 0], i1 = idx[n * 3 + 1], i2 = idx[n * 3 + 2];
    float w0 = wgt[n * 3 + 0], w1 = wgt[n * 3 + 1], w2 = wgt[n * 3 + 2];
    int Kc = C1 + C2;
    for (int c = threadIdx.x; c < C2; c += blockDim.x) {
        float v = w0 * kfeat[(size_t)i0 * C2 + c]
                + w1 * kfeat[(size_t)i1 * C2 + c]
                + w2 * kfeat[(size_t)i2 * C2 + c];
        X[(size_t)n * Kc + c] = (bf16)v;
    }
    for (int c = threadIdx.x; c < C1; c += blockDim.x) {
        X[(size_t)n * Kc + C2 + c] = (bf16)ufeat[(size_t)n * C1 + c];
    }
}

// ---------------------------------------------------------------------------
// Kernel 3: f32 -> bf16 conversion (weights)
// ---------------------------------------------------------------------------
__global__ void cvt_bf16_kernel(const float* __restrict__ w, bf16* __restrict__ wb, int n)
{
    int i = blockIdx.x * blockDim.x + threadIdx.x;
    if (i < n) wb[i] = (bf16)w[i];
}

// ---------------------------------------------------------------------------
// Kernel 4: WMMA GEMM + fused BN partial reduction
//   H[N,Cout] = X[N,K] * Wt[Cout,K]^T  (bf16 in, f32 acc)
//   stats[c]      += sum_n  H[n,c]   (per-block LDS reduce -> 1 atomic/col)
//   stats[C + c]  += sum_n  H[n,c]^2
//
// Block: 256 threads (8 waves) -> 256 rows x 64 cols.
// Each wave: 2 row-tiles x 4 col-tiles = 8 v8f accumulators; each B fragment
// (2x ds_load_b128) feeds 2 independent WMMAs.
//
// Fragment layouts per ISA 7.12.2 (16-bit):
//   A (16x32): lane half splits K into {kb..kb+7, kb+16..kb+23}, kb = half*8
//   B (32x16): lane holds column (lane&15), same K split as A (kb = half*8)
// ---------------------------------------------------------------------------
__global__ void gemm_wmma_kernel(const bf16* __restrict__ X,
                                 const bf16* __restrict__ Wt,
                                 float* __restrict__ H,
                                 float* __restrict__ stats,
                                 int N, int K, int Cout)
{
    extern __shared__ __align__(16) char smem_raw[];
    bf16* sW = (bf16*)smem_raw;           // [64][K + 8]
    __shared__ float sSum[64];
    __shared__ float sSqs[64];
    const int LK = K + 8;

    const int tid  = threadIdx.x;
    const int wave = tid >> 5;
    const int lane = tid & 31;
    const int half = lane >> 4;
    const int r    = lane & 15;

    const int rowBase = blockIdx.x * 256 + wave * 32;
    const int colBase = blockIdx.y * 64;

    // cooperative stage of the 64-column weight tile into LDS (b128 loads)
    const int kv = K >> 3;                 // 8 bf16 per 16B chunk
    for (int e = tid; e < 64 * kv; e += blockDim.x) {
        int co = e / kv;
        int kc = (e - co * kv) << 3;
        *(uint4*)&sW[co * LK + kc] = *(const uint4*)&Wt[(size_t)(colBase + co) * K + kc];
    }
    if (tid < 64) { sSum[tid] = 0.f; sSqs[tid] = 0.f; }
    __syncthreads();

    v8f acc[2][4];
#pragma unroll
    for (int t = 0; t < 2; ++t)
#pragma unroll
        for (int s = 0; s < 4; ++s) acc[t][s] = 0;

    const int aRow0 = min(rowBase + r,      N - 1);
    const int aRow1 = min(rowBase + 16 + r, N - 1);
    const bf16* aBase0 = X + (size_t)aRow0 * K + half * 8;
    const bf16* aBase1 = X + (size_t)aRow1 * K + half * 8;

    for (int k0 = 0; k0 < K; k0 += 32) {
        union { v16bf v; uint4 q[2]; } a0, a1, b;
        a0.q[0] = *(const uint4*)(aBase0 + k0);
        a0.q[1] = *(const uint4*)(aBase0 + k0 + 16);
        a1.q[0] = *(const uint4*)(aBase1 + k0);
        a1.q[1] = *(const uint4*)(aBase1 + k0 + 16);

        const bf16* bCol = sW + k0 + half * 8;    // K {kb..kb+7, kb+16..kb+23}, kb = half*8
#pragma unroll
        for (int s = 0; s < 4; ++s) {
            const bf16* bp = bCol + (s * 16 + r) * LK;
            b.q[0] = *(const uint4*)(bp);
            b.q[1] = *(const uint4*)(bp + 16);
            acc[0][s] = __builtin_amdgcn_wmma_f32_16x16x32_bf16(
                false, a0.v, false, b.v, (short)0, acc[0][s], false, false);
            acc[1][s] = __builtin_amdgcn_wmma_f32_16x16x32_bf16(
                false, a1.v, false, b.v, (short)0, acc[1][s], false, false);
        }
    }

    // store + per-lane BN partials
    // D layout: VGPR i, lanes 0-15 -> row i, lanes 16-31 -> row i+8; col = lane&15
#pragma unroll
    for (int s = 0; s < 4; ++s) {
        int colL = s * 16 + r;
        int col  = colBase + colL;
        float ps = 0.f, pq = 0.f;
#pragma unroll
        for (int t = 0; t < 2; ++t) {
#pragma unroll
            for (int i = 0; i < 8; ++i) {
                int row = rowBase + t * 16 + i + 8 * half;
                float v = acc[t][s][i];
                if (row < N) {
                    H[(size_t)row * Cout + col] = v;
                    ps += v;
                    pq += v * v;
                }
            }
        }
        unsafeAtomicAdd(&sSum[colL], ps);   // ds_add_f32
        unsafeAtomicAdd(&sSqs[colL], pq);
    }
    __syncthreads();
    if (tid < 64) {
        unsafeAtomicAdd(&stats[colBase + tid],        sSum[tid]);  // global_atomic_add_f32
        unsafeAtomicAdd(&stats[Cout + colBase + tid], sSqs[tid]);
    }
}

// ---------------------------------------------------------------------------
// BN helpers
// ---------------------------------------------------------------------------
__global__ void zero_kernel(float* __restrict__ p, int n)
{
    int i = blockIdx.x * blockDim.x + threadIdx.x;
    if (i < n) p[i] = 0.f;
}

template <bool OUT_BF16>
__global__ void bn_apply_kernel(const float* __restrict__ H,
                                const float* __restrict__ stats,
                                const float* __restrict__ gamma,
                                const float* __restrict__ beta,
                                int N, int C,
                                bf16* __restrict__ out_bf,
                                float* __restrict__ out_f)
{
    int n = blockIdx.x;
    int c = threadIdx.x;
    float invN = 1.0f / (float)N;
    float mean = stats[c] * invN;
    float var  = stats[C + c] * invN - mean * mean;
    float sc   = rsqrtf(var + 1e-5f) * gamma[c];
    float v    = (H[(size_t)n * C + c] - mean) * sc + beta[c];
    v = fmaxf(v, 0.0f);
    if (OUT_BF16) out_bf[(size_t)n * C + c] = (bf16)v;
    else          out_f [(size_t)n * C + c] = v;
}

// ---------------------------------------------------------------------------
// Host-side orchestration
// ---------------------------------------------------------------------------
static inline size_t alignUp(size_t x) { return (x + 255) & ~(size_t)255; }

extern "C" void kernel_launch(void* const* d_in, const int* in_sizes, int n_in,
                              void* d_out, int out_size, void* d_ws, size_t ws_size,
                              hipStream_t stream)
{
    const float* unknown = (const float*)d_in[0];
    const int*   ucnt    = (const int*)  d_in[1];
    const float* known   = (const float*)d_in[2];
    const int*   kcnt    = (const int*)  d_in[3];
    const float* ufeat   = (const float*)d_in[4];
    const float* kfeat   = (const float*)d_in[5];
    const float* W1      = (const float*)d_in[6];
    const float* g1      = (const float*)d_in[7];
    const float* b1      = (const float*)d_in[8];
    const float* W2      = (const float*)d_in[9];
    const float* g2      = (const float*)d_in[10];
    const float* b2      = (const float*)d_in[11];

    const int N   = in_sizes[0] / 3;
    const int M   = in_sizes[2] / 3;
    const int B   = in_sizes[1];
    const int C1  = in_sizes[4] / N;        // 128
    const int C2  = in_sizes[5] / M;        // 256
    const int K1  = C1 + C2;                // 384
    const int Co1 = in_sizes[6] / K1;       // 256
    const int Co2 = in_sizes[9] / Co1;      // 256
    const int CoMax = (Co1 > Co2) ? Co1 : Co2;

    // workspace carve
    char* ws = (char*)d_ws;
    int*   idx   = (int*)ws;            ws += alignUp((size_t)N * 3 * sizeof(int));
    float* wgt   = (float*)ws;          ws += alignUp((size_t)N * 3 * sizeof(float));
    bf16*  X     = (bf16*)ws;           ws += alignUp((size_t)N * K1 * sizeof(bf16));   // X1 then X2 (K1 >= Co1)
    float* H     = (float*)ws;          ws += alignUp((size_t)N * CoMax * sizeof(float));
    bf16*  Wb    = (bf16*)ws;           ws += alignUp((size_t)Co1 * K1 * sizeof(bf16)); // W1 then W2
    float* stats = (float*)ws;          ws += alignUp((size_t)2 * CoMax * sizeof(float));

    // 1) 3-NN + weights
    nn_kernel<<<(N + 255) / 256, 256, 0, stream>>>(unknown, ucnt, B, known, kcnt,
                                                   N, M, idx, wgt);

    // 2) interpolate + concat -> bf16 X1 [N, K1]
    interp_concat_kernel<<<N, 256, 0, stream>>>(idx, wgt, kfeat, ufeat, X, C1, C2);

    // 3) layer 1: W1 -> bf16, WMMA GEMM (fused BN reduce), BN apply -> bf16 X2
    cvt_bf16_kernel<<<(Co1 * K1 + 255) / 256, 256, 0, stream>>>(W1, Wb, Co1 * K1);
    zero_kernel<<<(2 * Co1 + 255) / 256, 256, 0, stream>>>(stats, 2 * Co1);
    {
        dim3 grid((N + 255) / 256, Co1 / 64);
        size_t lds = (size_t)64 * (K1 + 8) * sizeof(bf16);
        gemm_wmma_kernel<<<grid, 256, lds, stream>>>(X, Wb, H, stats, N, K1, Co1);
    }
    bn_apply_kernel<true><<<N, Co1, 0, stream>>>(H, stats, g1, b1, N, Co1, X, nullptr);

    // 4) layer 2: W2 -> bf16, WMMA GEMM (fused BN reduce), BN apply -> f32 out
    cvt_bf16_kernel<<<(Co2 * Co1 + 255) / 256, 256, 0, stream>>>(W2, Wb, Co2 * Co1);
    zero_kernel<<<(2 * Co2 + 255) / 256, 256, 0, stream>>>(stats, 2 * Co2);
    {
        dim3 grid((N + 255) / 256, Co2 / 64);
        size_t lds = (size_t)64 * (Co1 + 8) * sizeof(bf16);
        gemm_wmma_kernel<<<grid, 256, lds, stream>>>(X, Wb, H, stats, N, Co1, Co2);
    }
    bn_apply_kernel<false><<<N, Co2, 0, stream>>>(H, stats, g2, b2, N, Co2,
                                                  nullptr, (float*)d_out);
}
